// _FullBandLinearModule_17927193493572
// MI455X (gfx1250) — matrix operand order repaired and checked by reading, not compiled, not run on hardware
//
#include <hip/hip_runtime.h>

// MI455X (gfx1250) implementation of LN -> reduce(192->16)+SiLU ->
// 16x batched [F,F] full-band matmul -> expand(16->192)+SiLU -> residual.
//
// HBM-bound problem (~1.26 GB moved, ~10.5 GFLOP). Strategy:
//  k1: LN + reduce GEMM via v_wmma_f32_16x16x32_f16, writes Xt[n][p][f(pad288)] f16
//  k2: per-p GEMM  Y_p[n,g] = X_p[n,f] * f_W[p,g,f]^T  batched over 16 n-rows
//      so f_W (4.2MB, L2-resident) is streamed 125x instead of 2000x
//  k3: expand GEMM + SiLU + residual, one WMMA per 16x16 output tile (K=16 padded to 32)

typedef __attribute__((ext_vector_type(16))) _Float16 v16h;
typedef __attribute__((ext_vector_type(8)))  _Float16 v8h;
typedef __attribute__((ext_vector_type(8)))  float    v8f;

#define NB   2000
#define FF   257
#define CC   192
#define PP   16
#define FPAD 288                 // F padded to 9 K-chunks of 32
#define ROWS (NB * FF)           // 514000, divisible by 16
#define RT   (ROWS / 16)         // 32125 row tiles
#define NT   125                 // NB/16 n-tiles
#define GT   17                  // ceil(FF/16) g-tiles
#define CT   12                  // CC/16 c-tiles
#define K2JOBS (PP * NT * GT)    // 34000
#define K3JOBS (RT * CT)         // 385500

__device__ __forceinline__ float silu_f(float v) {
    return v * (1.0f / (1.0f + __expf(-v)));
}

// A-fragment K index for 16-bit 16x32 A (ISA 7.12.2): lane holds M=lane%16.
__device__ __forceinline__ int a_k(int j, int hsel) {
    int v = j >> 1, odd = j & 1;
    return ((v >> 2) << 4) + (hsel ? 8 : 0) + ((v & 3) << 1) + odd;
}
// Contiguous K index (k = 16*hsel + j) for 16-bit 32x16 B, not AMD's table order; unused: N=lane%16, K groups split at lane 16.
__device__ __forceinline__ int b_k(int j, int hsel) {
    int v = j >> 1, odd = j & 1;
    return (hsel ? 16 : 0) + (v << 1) + odd;
}

// ---------------- Kernel 1: LayerNorm + reduce GEMM + SiLU ----------------
__global__ __launch_bounds__(256) void k1_ln_reduce(
    const float* __restrict__ h, const float* __restrict__ ln_w,
    const float* __restrict__ ln_b, const float* __restrict__ red_W,
    const float* __restrict__ red_b, _Float16* __restrict__ Xt)
{
    const int lane = threadIdx.x & 31;
    const int wid  = threadIdx.x >> 5;
    const int tile = blockIdx.x * 8 + wid;       // wave-uniform
    if (tile >= RT) return;
    const int row0 = tile * 16;
    const int hsel = (lane >= 16) ? 1 : 0;

    // LN stats: each lane keeps mean/rstd only for the row it owns in A (row0 + lane%16)
    float myMu = 0.f, myRs = 0.f;
    for (int r = 0; r < 16; ++r) {
        const float* hr = h + (size_t)(row0 + r) * CC;
        float s = 0.f, s2 = 0.f;
        #pragma unroll
        for (int i = 0; i < 6; ++i) {
            float v = hr[lane + 32 * i];
            s += v; s2 += v * v;
        }
        #pragma unroll
        for (int m = 16; m >= 1; m >>= 1) {
            s  += __shfl_xor(s,  m, 32);
            s2 += __shfl_xor(s2, m, 32);
        }
        float mu  = s * (1.0f / CC);
        float var = s2 * (1.0f / CC) - mu * mu;
        float rs  = rsqrtf(var + 1e-5f);
        if (r == (lane & 15)) { myMu = mu; myRs = rs; }
    }

    const int   myRow = row0 + (lane & 15);
    const float* hrow = h + (size_t)myRow * CC;
    const int   pcol  = lane & 15;

    v8f acc = {};
    #pragma unroll
    for (int kc = 0; kc < 6; ++kc) {          // C=192 -> 6 K-chunks of 32
        const int cb = kc * 32;
        v16h a, b;
        #pragma unroll
        for (int j = 0; j < 16; ++j) {
            int c = cb + a_k(j, hsel);
            float x = (hrow[c] - myMu) * myRs * ln_w[c] + ln_b[c];
            a[j] = (_Float16)x;
            b[j] = (_Float16)red_W[pcol * CC + cb + a_k(j, hsel)];
        }
        acc = __builtin_amdgcn_wmma_f32_16x16x32_f16(false, a, false, b,
                                                     (short)0, acc, false, false);
    }
    // D layout: VGPR r -> row M = r + 8*hsel, col N = lane%16 = p
    #pragma unroll
    for (int r = 0; r < 8; ++r) {
        int   rowg = row0 + r + (hsel ? 8 : 0);
        int   n = rowg / FF;
        int   f = rowg - n * FF;
        float v = acc[r] + red_b[pcol];
        Xt[((size_t)n * PP + pcol) * FPAD + f] = (_Float16)silu_f(v);
    }
}

// ------------- Kernel 2: per-p full-band GEMM (Y = X * f_W^T + f_b) -------------
__global__ __launch_bounds__(256) void k2_fullband(
    const _Float16* __restrict__ Xt, const float* __restrict__ f_W,
    const float* __restrict__ f_b, _Float16* __restrict__ Y)
{
    const int lane = threadIdx.x & 31;
    const int wid  = threadIdx.x >> 5;
    const int job  = blockIdx.x * 8 + wid;        // wave-uniform
    if (job >= K2JOBS) return;
    const int p  = job / (NT * GT);
    const int rm = job - p * (NT * GT);
    const int nt = rm / GT;
    const int gt = rm - nt * GT;
    const int hsel = (lane >= 16) ? 1 : 0;

    const int n = nt * 16 + (lane & 15);                          // A row (M)
    const _Float16* xrow = Xt + ((size_t)n * PP + p) * FPAD;
    const int g   = gt * 16 + (lane & 15);                        // B col (N)
    const bool gok = (g < FF);
    const float* wrow = f_W + ((size_t)p * FF + (gok ? g : 0)) * FF;
    const int s = hsel ? 8 : 0;

    v8f acc = {};
    for (int kc = 0; kc < 9; ++kc) {             // F padded 288 -> 9 chunks of 32
        const int fb = kc * 32;
        // A: two contiguous 16B loads (Xt padded region is zeroed)
        v8h alo = *(const v8h*)(xrow + fb + s);
        v8h ahi = *(const v8h*)(xrow + fb + 16 + s);
        v16h a = __builtin_shufflevector(alo, ahi, 0,1,2,3,4,5,6,7,
                                                   8,9,10,11,12,13,14,15);
        v16h b;
        #pragma unroll
        for (int j = 0; j < 16; ++j) {
            int f2 = fb + a_k(j, hsel);
            b[j] = (gok && f2 < FF) ? (_Float16)wrow[f2] : (_Float16)0.f;
        }
        acc = __builtin_amdgcn_wmma_f32_16x16x32_f16(false, a, false, b,
                                                     (short)0, acc, false, false);
    }
    #pragma unroll
    for (int r = 0; r < 8; ++r) {
        if (gok) {
            int   nn = nt * 16 + r + s;
            float y  = acc[r] + f_b[p * FF + g];
            Y[((size_t)nn * FF + g) * PP + p] = (_Float16)y;
        }
    }
}

// ---------- Kernel 3: expand GEMM (K=16 pad 32) + SiLU + residual ----------
__global__ __launch_bounds__(256) void k3_expand(
    const _Float16* __restrict__ Y, const float* __restrict__ exp_W,
    const float* __restrict__ exp_b, const float* __restrict__ h,
    float* __restrict__ out)
{
    const int lane = threadIdx.x & 31;
    const int wid  = threadIdx.x >> 5;
    const int job  = blockIdx.x * 8 + wid;        // wave-uniform
    if (job >= K3JOBS) return;
    const int rt = job / CT;
    const int ct = job - rt * CT;
    const int hsel = (lane >= 16) ? 1 : 0;
    const int s = hsel ? 8 : 0;

    const int row = rt * 16 + (lane & 15);
    const _Float16* yrow = Y + (size_t)row * PP;
    const int c = ct * 16 + (lane & 15);

    // A: K=0..15 valid (one 16B load), K=16..31 zero padding
    v8h alo = *(const v8h*)(yrow + s);
    v8h zer = {};
    v16h a = __builtin_shufflevector(alo, zer, 0,1,2,3,4,5,6,7,
                                               8,9,10,11,12,13,14,15);
    v16h b;
    #pragma unroll
    for (int j = 0; j < 16; ++j) {
        int kb = a_k(j, hsel);
        b[j] = (kb < PP) ? (_Float16)exp_W[c * PP + kb] : (_Float16)0.f;
    }
    v8f acc = {};
    acc = __builtin_amdgcn_wmma_f32_16x16x32_f16(false, a, false, b,
                                                 (short)0, acc, false, false);
    #pragma unroll
    for (int r = 0; r < 8; ++r) {
        int    rowg = rt * 16 + r + s;
        size_t idx  = (size_t)rowg * CC + c;
        float  v    = acc[r] + exp_b[c];
        out[idx] = h[idx] + silu_f(v);
    }
}

extern "C" void kernel_launch(void* const* d_in, const int* in_sizes, int n_in,
                              void* d_out, int out_size, void* d_ws, size_t ws_size,
                              hipStream_t stream) {
    const float* h     = (const float*)d_in[0];
    const float* ln_w  = (const float*)d_in[1];
    const float* ln_b  = (const float*)d_in[2];
    const float* red_W = (const float*)d_in[3];
    const float* red_b = (const float*)d_in[4];
    const float* f_W   = (const float*)d_in[5];
    const float* f_b   = (const float*)d_in[6];
    const float* exp_W = (const float*)d_in[7];
    const float* exp_b = (const float*)d_in[8];
    float* out = (float*)d_out;

    // Scratch layout: Xt f16 [NB][PP][FPAD] (18.43 MB), then Y f16 [NB][FF][PP] (16.45 MB)
    _Float16* Xt = (_Float16*)d_ws;
    const size_t xbytes = (size_t)NB * PP * FPAD * sizeof(_Float16);
    _Float16* Yb = (_Float16*)((char*)d_ws + xbytes);

    // Zero Xt so the f in [257,288) K-padding reads as 0 in k2 (graph-capture safe)
    hipMemsetAsync(Xt, 0, xbytes, stream);

    dim3 blk(256);
    k1_ln_reduce<<<(RT + 7) / 8,     blk, 0, stream>>>(h, ln_w, ln_b, red_W, red_b, Xt);
    k2_fullband <<<(K2JOBS + 7) / 8, blk, 0, stream>>>(Xt, f_W, f_b, Yb);
    k3_expand   <<<(K3JOBS + 7) / 8, blk, 0, stream>>>(Yb, exp_W, exp_b, h, out);
}
